// EdgeModel_35012573397767
// MI455X (gfx1250) — matrix operand order repaired and checked by reading, not compiled, not run on hardware
//
#include <hip/hip_runtime.h>
#include <hip/hip_bf16.h>

typedef __attribute__((ext_vector_type(16))) __bf16 v16bf;
typedef __attribute__((ext_vector_type(8)))  float  v8f;

__device__ __forceinline__ unsigned short f2bf(float f) {
    union { float f; unsigned u; } c;
    c.f = f;
    unsigned u = c.u;
    u += 0x7FFFu + ((u >> 16) & 1u);           // round-to-nearest-even
    return (unsigned short)(u >> 16);
}

// ---------------------------------------------------------------------------
// Prep: pack W [768,256] f32 (row-major) into bf16-pair words:
//   Wpack[k2*256 + n] = { bf16(W[2*k2][n]) , bf16(W[2*k2+1][n]) }  (lo = even K)
// 384 KB total -> stays resident in the 192 MB L2 for the whole GEMM.
// ---------------------------------------------------------------------------
__global__ __launch_bounds__(256) void pack_w_kernel(const float* __restrict__ W,
                                                     unsigned* __restrict__ Wp) {
    int idx = blockIdx.x * 256 + threadIdx.x;   // 0 .. 384*256-1
    int k2  = idx >> 8;                          // 0..383
    int n   = idx & 255;
    float a = W[(2 * k2)     * 256 + n];
    float c = W[(2 * k2 + 1) * 256 + n];
    Wp[idx] = (unsigned)f2bf(a) | ((unsigned)f2bf(c) << 16);
}

// ---------------------------------------------------------------------------
// GEMM: out[E,256] = relu( X[E,768] @ W + b ),
// X = concat([dest(256) | src(256) | edge_attr(128) | u[batch](128)]).
// Tile: 64 rows x 256 cols per workgroup (8 wave32s). K chunk = 32 (bf16 WMMA).
// LDS X-tile row stride padded to 20 words (80B): keeps ds_load_b128 16B
// alignment while spreading lanes over all 64 banks (2-way ideal, no 4-way
// conflicts that a 16-word stride produces).
// ---------------------------------------------------------------------------
#define M_TILE 64
#define LROW   20   // LDS words per X-tile row (16 data + 4 pad)

__global__ __launch_bounds__(256) void edge_gemm_kernel(
    const float* __restrict__ src,  const float* __restrict__ dest,
    const float* __restrict__ edge, const float* __restrict__ u,
    const int*   __restrict__ batch, const unsigned* __restrict__ Wp,
    const float* __restrict__ bias, float* __restrict__ out, int E)
{
    __shared__ unsigned ldsX[2][M_TILE * LROW]; // bf16-pair words, double buffered
    __shared__ int shBatch[M_TILE];

    const int  tid  = threadIdx.x;
    const int  lane = tid & 31;
    const int  wave = tid >> 5;
    const long m0   = (long)blockIdx.x * M_TILE;

    if (tid < M_TILE) {
        long r = m0 + tid;
        if (r > (long)E - 1) r = (long)E - 1;   // clamp (E >= 1)
        shBatch[tid] = batch[r];
    }

    // Per-thread staging geometry is kt-invariant: precompute row bases.
    // Each thread owns 4 (m, kp) slots: idx = tid + it*256.
    long rowBase256[4];   // row*256 + 2*kp  (for dest/src sections)
    long rowBase128[4];   // row*128 + 2*kp  (for edge/u sections)
    int  slot[4];         // LDS word index (padded row stride)
    int  mrow[4];
    int  kp2[4];
    #pragma unroll
    for (int it = 0; it < 4; ++it) {
        int idx = tid + it * 256;               // 0..1023
        int m   = idx >> 4;                     // row in tile
        int kp  = idx & 15;                     // K-pair index in chunk
        long row = m0 + m;
        if (row > (long)E - 1) row = (long)E - 1;   // branchless clamp
        rowBase256[it] = row * 256 + 2 * kp;
        rowBase128[it] = row * 128 + 2 * kp;
        slot[it] = m * LROW + kp;
        mrow[it] = m;
        kp2[it]  = 2 * kp;
    }

    // Stage the 64x32 X chunk for k-tile `kt` into `buf` as packed bf16 pairs.
    // Phase 1: issue all global loads (clause-able, no divergence).
    // Phase 2: convert + LDS store.
    auto stage = [&](int kt, unsigned* buf) {
        float2 v[4];
        #pragma unroll
        for (int it = 0; it < 4; ++it) {
            const float* p;
            if      (kt <  8) p = dest + rowBase256[it] + kt * 32;
            else if (kt < 16) p = src  + rowBase256[it] + (kt -  8) * 32;
            else if (kt < 20) p = edge + rowBase128[it] + (kt - 16) * 32;
            else              p = u + (long)shBatch[mrow[it]] * 128
                                    + kp2[it] + (kt - 20) * 32;
            v[it] = *(const float2*)p;
        }
        #pragma unroll
        for (int it = 0; it < 4; ++it)
            buf[slot[it]] = (unsigned)f2bf(v[it].x)
                          | ((unsigned)f2bf(v[it].y) << 16);
    };

    v8f acc[4][2];
    #pragma unroll
    for (int mi = 0; mi < 4; ++mi)
        #pragma unroll
        for (int ni = 0; ni < 2; ++ni)
            acc[mi][ni] = (v8f){0.f, 0.f, 0.f, 0.f, 0.f, 0.f, 0.f, 0.f};

    const int nlo  = lane & 15;                 // N within 16-wide subtile
    const int hrow = (lane >= 16) ? 8 : 0;      // B: lane-half K split (pairs)
    const int arow = lane & 15;                 // A: M row
    const int aoff = (lane >= 16) ? 8 : 0;      // A: lane-half K split (pairs)

    stage(0, ldsX[0]);
    __syncthreads();

    for (int kt = 0; kt < 24; ++kt) {
        unsigned* cur = ldsX[kt & 1];
        if (kt + 1 < 24) stage(kt + 1, ldsX[(kt + 1) & 1]);

        // B fragments (2 x 16-wide N strips per wave), straight from L2.
        union BF { unsigned u[8]; v16bf v; } bfr[2];
        const unsigned* wb = Wp + (kt * 16 + hrow) * 256 + wave * 32 + nlo;
        #pragma unroll
        for (int ni = 0; ni < 2; ++ni)
            #pragma unroll
            for (int j = 0; j < 8; ++j)
                bfr[ni].u[j] = wb[j * 256 + ni * 16];

        // A fragments from LDS (2x ds_load_b128 each) + 8 WMMAs.
        #pragma unroll
        for (int mi = 0; mi < 4; ++mi) {
            union AF { uint4 q[2]; v16bf v; } af;
            const unsigned* ab = cur + (mi * 16 + arow) * LROW + aoff;
            af.q[0] = *(const uint4*)(ab);
            af.q[1] = *(const uint4*)(ab + 4);
            #pragma unroll
            for (int ni = 0; ni < 2; ++ni)
                acc[mi][ni] = __builtin_amdgcn_wmma_f32_16x16x32_bf16(
                    false, af.v, false, bfr[ni].v,
                    (short)0, acc[mi][ni], false, false);
        }
        __syncthreads();
    }

    // Epilogue: bias + ReLU + store (C/D layout: vgpr r -> M, lane half -> M+8).
    float bv[2];
    bv[0] = bias[wave * 32 + nlo];
    bv[1] = bias[wave * 32 + 16 + nlo];
    const int rowadd = (lane >= 16) ? 8 : 0;

    #pragma unroll
    for (int mi = 0; mi < 4; ++mi) {
        #pragma unroll
        for (int r = 0; r < 8; ++r) {
            long row = m0 + mi * 16 + rowadd + r;
            if (row < E) {
                #pragma unroll
                for (int ni = 0; ni < 2; ++ni) {
                    float v = acc[mi][ni][r] + bv[ni];
                    v = fmaxf(v, 0.f);
                    out[row * 256 + wave * 32 + ni * 16 + nlo] = v;
                }
            }
        }
    }
}

// ---------------------------------------------------------------------------
// d_in order (setup_inputs): 0=src, 1=dest, 2=edge_attr, 3=u, 4=batch, 5=W, 6=b
// ---------------------------------------------------------------------------
extern "C" void kernel_launch(void* const* d_in, const int* in_sizes, int n_in,
                              void* d_out, int out_size, void* d_ws, size_t ws_size,
                              hipStream_t stream) {
    const float* src   = (const float*)d_in[0];
    const float* dest  = (const float*)d_in[1];
    const float* edge  = (const float*)d_in[2];
    const float* u     = (const float*)d_in[3];
    const int*   batch = (const int*)  d_in[4];
    const float* W     = (const float*)d_in[5];
    const float* b     = (const float*)d_in[6];
    float*       out   = (float*)d_out;
    unsigned*    Wp    = (unsigned*)d_ws;       // 384*256*4 = 393216 bytes

    const int E = in_sizes[4];                  // batch has one entry per edge

    pack_w_kernel<<<384, 256, 0, stream>>>(W, Wp);

    const int nblocks = (E + M_TILE - 1) / M_TILE;
    edge_gemm_kernel<<<nblocks, 256, 0, stream>>>(src, dest, edge, u, batch,
                                                  Wp, b, out, E);
}
